// MultiHeadAttention_90005334655147
// MI455X (gfx1250) — matrix operand order repaired and checked by reading, not compiled, not run on hardware
//
#include <hip/hip_runtime.h>
#include <hip/hip_bf16.h>

typedef __attribute__((ext_vector_type(16))) __bf16 v16bf;
typedef __attribute__((ext_vector_type(8)))  __bf16 v8bf;
typedef __attribute__((ext_vector_type(8)))  float  v8f;

#define S_LEN   2048
#define D_MODEL 512
#define NH      8
#define DKV     64
#define B_BATCH 2
#define M_ROWS  (B_BATCH * S_LEN)   // 4096

// ================= fragment loaders (wave32, gfx1250 ISA 7.12.2 layouts) =================
// A 16x32 bf16 from row-major [.., M, K-contig]: lane = row (lane&15);
//   lanes 0-15 hold K={0..7,16..23}, lanes 16-31 K={8..15,24..31} -> two contiguous 16B runs.
__device__ __forceinline__ v16bf load_a_bf16(const __bf16* __restrict__ base, int ld,
                                             int m0, int k0) {
    const int lane = threadIdx.x & 31;
    const __bf16* p = base + (size_t)(m0 + (lane & 15)) * ld + k0 + ((lane < 16) ? 0 : 8);
    v8bf lo = *(const v8bf*)p;
    v8bf hi = *(const v8bf*)(p + 16);
    v16bf a;
#pragma unroll
    for (int i = 0; i < 8; ++i) { a[i] = lo[i]; a[i + 8] = hi[i]; }
    return a;
}

// Same A fragment but from an f32 row-major source with convert (v_cvt_pk_bf16_f32).
__device__ __forceinline__ v16bf load_a_f32(const float* __restrict__ base, int ld,
                                            int m0, int k0) {
    const int lane = threadIdx.x & 31;
    const float* p = base + (size_t)(m0 + (lane & 15)) * ld + k0 + ((lane < 16) ? 0 : 8);
    float4 x0 = ((const float4*)p)[0],        x1 = ((const float4*)p)[1];
    float4 y0 = ((const float4*)(p + 16))[0], y1 = ((const float4*)(p + 16))[1];
    v16bf a;
    a[0]=(__bf16)x0.x; a[1]=(__bf16)x0.y; a[2]=(__bf16)x0.z; a[3]=(__bf16)x0.w;
    a[4]=(__bf16)x1.x; a[5]=(__bf16)x1.y; a[6]=(__bf16)x1.z; a[7]=(__bf16)x1.w;
    a[8]=(__bf16)y0.x; a[9]=(__bf16)y0.y; a[10]=(__bf16)y0.z; a[11]=(__bf16)y0.w;
    a[12]=(__bf16)y1.x; a[13]=(__bf16)y1.y; a[14]=(__bf16)y1.z; a[15]=(__bf16)y1.w;
    return a;
}

// B 32x16 bf16 from N-major [.., N, K-contig]: lane = col (lane&15);
//   lanes 0-15 hold K={0..7,16..23}, lanes 16-31 K={8..15,24..31} -> two contiguous 16B runs.
__device__ __forceinline__ v16bf load_b_bf16(const __bf16* __restrict__ base, int ld,
                                             int n0, int k0) {
    const int lane = threadIdx.x & 31;
    const __bf16* p = base + (size_t)(n0 + (lane & 15)) * ld + k0 + ((lane < 16) ? 0 : 8);
    v8bf lo = *(const v8bf*)p;
    v8bf hi = *(const v8bf*)(p + 16);
    v16bf b;
#pragma unroll
    for (int i = 0; i < 8; ++i) { b[i] = lo[i]; b[i + 8] = hi[i]; }
    return b;
}

__device__ __forceinline__ v8f wmma_bf16(v16bf a, v16bf b, v8f c) {
    return __builtin_amdgcn_wmma_f32_16x16x32_bf16(false, a, false, b, (short)0, c, false, false);
}

// C/D 16x16 f32: N = lane&15, M = r + ((lane<16)?0:8)
__device__ __forceinline__ void cd_coords(int& c, int& rbase) {
    const int lane = threadIdx.x & 31;
    c = lane & 15;
    rbase = (lane < 16) ? 0 : 8;
}

// ============ 0) one-time weight transpose+convert: W[K=512][N=512] -> WT[N][K] bf16 ============
__global__ __launch_bounds__(256)
void prep_wt_kernel(const float* __restrict__ W, __bf16* __restrict__ WT)
{
    const int idx = blockIdx.x * 256 + threadIdx.x;   // over 512*512, k fastest in output
    const int n = idx >> 9, k = idx & 511;
    WT[idx] = (__bf16)W[k * D_MODEL + n];
}

// ============ 1) projection: Y = X @ W + b, wave = 32(M) x 64(N) tile ============
// transposeOut==0 -> Y[b,h,s,dk] row-major   (qh, kh)
// transposeOut==1 -> Y[b,h,dk,s] N-major     (vhT)
__global__ __launch_bounds__(32)
void proj_kernel(const float* __restrict__ X, const __bf16* __restrict__ WT,
                 const float* __restrict__ bias, __bf16* __restrict__ Y, int transposeOut)
{
    const int n0 = blockIdx.x * 64;
    const int m0 = blockIdx.y * 32;
    v8f acc[8] = {};
#pragma unroll 2
    for (int k0 = 0; k0 < D_MODEL; k0 += 32) {
        v16bf a0 = load_a_f32(X, D_MODEL, m0,      k0);
        v16bf a1 = load_a_f32(X, D_MODEL, m0 + 16, k0);
        v16bf b0 = load_b_bf16(WT, D_MODEL, n0 +  0, k0);
        v16bf b1 = load_b_bf16(WT, D_MODEL, n0 + 16, k0);
        v16bf b2 = load_b_bf16(WT, D_MODEL, n0 + 32, k0);
        v16bf b3 = load_b_bf16(WT, D_MODEL, n0 + 48, k0);
        acc[0] = wmma_bf16(a0, b0, acc[0]);  acc[1] = wmma_bf16(a0, b1, acc[1]);
        acc[2] = wmma_bf16(a0, b2, acc[2]);  acc[3] = wmma_bf16(a0, b3, acc[3]);
        acc[4] = wmma_bf16(a1, b0, acc[4]);  acc[5] = wmma_bf16(a1, b1, acc[5]);
        acc[6] = wmma_bf16(a1, b2, acc[6]);  acc[7] = wmma_bf16(a1, b3, acc[7]);
    }
    int c, rbase; cd_coords(c, rbase);
    const int h  = n0 >> 6;          // n-strip is 64-aligned: one head per wave
    const int bi = m0 >> 11;
#pragma unroll
    for (int mt = 0; mt < 2; ++mt) {
#pragma unroll
        for (int nt = 0; nt < 4; ++nt) {
            const int n  = n0 + nt * 16 + c;
            const int dk = n & 63;
            const float bv = bias[n];
#pragma unroll
            for (int r = 0; r < 8; ++r) {
                const int s = (m0 + mt * 16 + rbase + r) & (S_LEN - 1);
                const float val = acc[mt * 4 + nt][r] + bv;
                size_t idx = transposeOut
                    ? ((size_t)((bi * NH + h) * DKV + dk) * S_LEN + s)
                    : ((size_t)((bi * NH + h) * S_LEN + s) * DKV + dk);
                Y[idx] = (__bf16)val;
            }
        }
    }
}

// ============ 2) scores = qh @ kh^T / 8, mask; wave = 32(q) x 64(key) ============
__global__ __launch_bounds__(32)
void qk_kernel(const __bf16* __restrict__ qh, const __bf16* __restrict__ kh,
               const unsigned char* __restrict__ mask, float* __restrict__ attn)
{
    const int kk0 = blockIdx.x * 64;
    const int q0  = blockIdx.y * 32;
    const int bh  = blockIdx.z;
    const int bi  = bh >> 3;
    const __bf16* qbase = qh + (size_t)bh * S_LEN * DKV;
    const __bf16* kbase = kh + (size_t)bh * S_LEN * DKV;   // key-major, dk contiguous => B frags
    v8f acc[8] = {};
#pragma unroll
    for (int k0 = 0; k0 < DKV; k0 += 32) {
        v16bf a0 = load_a_bf16(qbase, DKV, q0,      k0);
        v16bf a1 = load_a_bf16(qbase, DKV, q0 + 16, k0);
        v16bf b0 = load_b_bf16(kbase, DKV, kk0 +  0, k0);
        v16bf b1 = load_b_bf16(kbase, DKV, kk0 + 16, k0);
        v16bf b2 = load_b_bf16(kbase, DKV, kk0 + 32, k0);
        v16bf b3 = load_b_bf16(kbase, DKV, kk0 + 48, k0);
        acc[0] = wmma_bf16(a0, b0, acc[0]);  acc[1] = wmma_bf16(a0, b1, acc[1]);
        acc[2] = wmma_bf16(a0, b2, acc[2]);  acc[3] = wmma_bf16(a0, b3, acc[3]);
        acc[4] = wmma_bf16(a1, b0, acc[4]);  acc[5] = wmma_bf16(a1, b1, acc[5]);
        acc[6] = wmma_bf16(a1, b2, acc[6]);  acc[7] = wmma_bf16(a1, b3, acc[7]);
    }
    int c, rbase; cd_coords(c, rbase);
    float* obase = attn + (size_t)bh * S_LEN * S_LEN;
    const unsigned char* mbase = mask + (size_t)bi * S_LEN * S_LEN;
#pragma unroll
    for (int mt = 0; mt < 2; ++mt) {
#pragma unroll
        for (int nt = 0; nt < 4; ++nt) {
#pragma unroll
            for (int r = 0; r < 8; ++r) {
                const int q  = q0 + mt * 16 + rbase + r;
                const int kk = kk0 + nt * 16 + c;
                float val = acc[mt * 4 + nt][r] * 0.125f;
                if (mbase[(size_t)q * S_LEN + kk]) val = -__builtin_inff();
                obase[(size_t)q * S_LEN + kk] = val;
            }
        }
    }
}

// ============ 3) row softmax over 2048, in place ============
__global__ __launch_bounds__(256)
void softmax_kernel(float* __restrict__ attn)
{
    float* p = attn + (size_t)blockIdx.x * S_LEN;
    __shared__ float red[256];
    const int t = threadIdx.x;
    float vals[8];
    float lmax = -__builtin_inff();
#pragma unroll
    for (int i = 0; i < 8; ++i) { vals[i] = p[t + i * 256]; lmax = fmaxf(lmax, vals[i]); }
    red[t] = lmax; __syncthreads();
    for (int s = 128; s > 0; s >>= 1) { if (t < s) red[t] = fmaxf(red[t], red[t + s]); __syncthreads(); }
    const float m = red[0]; __syncthreads();
    float lsum = 0.f;
#pragma unroll
    for (int i = 0; i < 8; ++i) { vals[i] = expf(vals[i] - m); lsum += vals[i]; }
    red[t] = lsum; __syncthreads();
    for (int s = 128; s > 0; s >>= 1) { if (t < s) red[t] += red[t + s]; __syncthreads(); }
    const float inv = 1.0f / red[0];
#pragma unroll
    for (int i = 0; i < 8; ++i) p[t + i * 256] = vals[i] * inv;
}

// ============ 4) ctx = attn @ vh; wave = 16(q) x 64(dv) (full DV) ============
__global__ __launch_bounds__(32)
void pv_kernel(const float* __restrict__ attn, const __bf16* __restrict__ vhT,
               __bf16* __restrict__ ctx)
{
    const int q0 = blockIdx.x * 16;
    const int bh = blockIdx.y;
    const int bi = bh >> 3, h = bh & 7;
    const float*  pbase = attn + (size_t)bh * S_LEN * S_LEN;
    const __bf16* vbase = vhT  + (size_t)bh * DKV * S_LEN;  // dv-major, key contiguous => B frags
    v8f acc[4] = {};
#pragma unroll 2
    for (int k0 = 0; k0 < S_LEN; k0 += 32) {
        v16bf a  = load_a_f32(pbase, S_LEN, q0, k0);
        v16bf b0 = load_b_bf16(vbase, S_LEN,  0, k0);
        v16bf b1 = load_b_bf16(vbase, S_LEN, 16, k0);
        v16bf b2 = load_b_bf16(vbase, S_LEN, 32, k0);
        v16bf b3 = load_b_bf16(vbase, S_LEN, 48, k0);
        acc[0] = wmma_bf16(a, b0, acc[0]);  acc[1] = wmma_bf16(a, b1, acc[1]);
        acc[2] = wmma_bf16(a, b2, acc[2]);  acc[3] = wmma_bf16(a, b3, acc[3]);
    }
    int c, rbase; cd_coords(c, rbase);
#pragma unroll
    for (int nt = 0; nt < 4; ++nt) {
#pragma unroll
        for (int r = 0; r < 8; ++r) {
            const int q = q0 + rbase + r;
            ctx[(size_t)(bi * S_LEN + q) * D_MODEL + h * DKV + nt * 16 + c] = (__bf16)acc[nt][r];
        }
    }
}

// ============ 5) X = ctx @ Wfc + bfc + residual; wave = 32 x 64 ============
__global__ __launch_bounds__(32)
void fc_kernel(const __bf16* __restrict__ ctx, const __bf16* __restrict__ WfcT,
               const float* __restrict__ bfc, const float* __restrict__ resid,
               float* __restrict__ Xout)
{
    const int n0 = blockIdx.x * 64;
    const int m0 = blockIdx.y * 32;
    v8f acc[8] = {};
#pragma unroll 2
    for (int k0 = 0; k0 < D_MODEL; k0 += 32) {
        v16bf a0 = load_a_bf16(ctx, D_MODEL, m0,      k0);
        v16bf a1 = load_a_bf16(ctx, D_MODEL, m0 + 16, k0);
        v16bf b0 = load_b_bf16(WfcT, D_MODEL, n0 +  0, k0);
        v16bf b1 = load_b_bf16(WfcT, D_MODEL, n0 + 16, k0);
        v16bf b2 = load_b_bf16(WfcT, D_MODEL, n0 + 32, k0);
        v16bf b3 = load_b_bf16(WfcT, D_MODEL, n0 + 48, k0);
        acc[0] = wmma_bf16(a0, b0, acc[0]);  acc[1] = wmma_bf16(a0, b1, acc[1]);
        acc[2] = wmma_bf16(a0, b2, acc[2]);  acc[3] = wmma_bf16(a0, b3, acc[3]);
        acc[4] = wmma_bf16(a1, b0, acc[4]);  acc[5] = wmma_bf16(a1, b1, acc[5]);
        acc[6] = wmma_bf16(a1, b2, acc[6]);  acc[7] = wmma_bf16(a1, b3, acc[7]);
    }
    int c, rbase; cd_coords(c, rbase);
#pragma unroll
    for (int mt = 0; mt < 2; ++mt) {
#pragma unroll
        for (int nt = 0; nt < 4; ++nt) {
            const int n = n0 + nt * 16 + c;
            const float bv = bfc[n];
#pragma unroll
            for (int r = 0; r < 8; ++r) {
                const int m = m0 + mt * 16 + rbase + r;
                Xout[(size_t)m * D_MODEL + n] = acc[mt * 4 + nt][r] + bv + resid[(size_t)m * D_MODEL + n];
            }
        }
    }
}

// ============ 6) layernorm over D=512 per row ============
__global__ __launch_bounds__(256)
void ln_kernel(const float* __restrict__ X, const float* __restrict__ gamma,
               const float* __restrict__ beta, float* __restrict__ out)
{
    const float* p = X + (size_t)blockIdx.x * D_MODEL;
    float* o = out + (size_t)blockIdx.x * D_MODEL;
    __shared__ float rs[256], rs2[256];
    const int t = threadIdx.x;
    float x0 = p[t], x1 = p[t + 256];
    rs[t]  = x0 + x1;
    rs2[t] = x0 * x0 + x1 * x1;
    __syncthreads();
    for (int s = 128; s > 0; s >>= 1) {
        if (t < s) { rs[t] += rs[t + s]; rs2[t] += rs2[t + s]; }
        __syncthreads();
    }
    const float mu   = rs[0] * (1.0f / D_MODEL);
    const float var  = rs2[0] * (1.0f / D_MODEL) - mu * mu;
    const float rstd = rsqrtf(var + 1e-5f);
    o[t]       = (x0 - mu) * rstd * gamma[t]       + beta[t];
    o[t + 256] = (x1 - mu) * rstd * gamma[t + 256] + beta[t + 256];
}

// ================= launch =================
extern "C" void kernel_launch(void* const* d_in, const int* in_sizes, int n_in,
                              void* d_out, int out_size, void* d_ws, size_t ws_size,
                              hipStream_t stream)
{
    const float* q     = (const float*)d_in[0];
    const float* k     = (const float*)d_in[1];
    const float* v     = (const float*)d_in[2];
    const unsigned char* mask = (const unsigned char*)d_in[3];
    const float* Wq    = (const float*)d_in[4];
    const float* bq    = (const float*)d_in[5];
    const float* Wk    = (const float*)d_in[6];
    const float* bk    = (const float*)d_in[7];
    const float* Wfc   = (const float*)d_in[8];
    const float* bfc   = (const float*)d_in[9];
    const float* gamma = (const float*)d_in[10];
    const float* beta  = (const float*)d_in[11];

    float* out_final = (float*)d_out;
    float* attn = out_final + (size_t)B_BATCH * S_LEN * D_MODEL;   // second output

    const size_t we = (size_t)D_MODEL * D_MODEL;   // 512*512
    const size_t pe = (size_t)M_ROWS * D_MODEL;    // 4096*512
    char* ws = (char*)d_ws;
    __bf16* WqT  = (__bf16*)ws;  ws += we * sizeof(__bf16);
    __bf16* WkT  = (__bf16*)ws;  ws += we * sizeof(__bf16);
    __bf16* WfcT = (__bf16*)ws;  ws += we * sizeof(__bf16);
    __bf16* qh   = (__bf16*)ws;  ws += pe * sizeof(__bf16);
    __bf16* kh   = (__bf16*)ws;  ws += pe * sizeof(__bf16);
    __bf16* vhT  = (__bf16*)ws;  ws += pe * sizeof(__bf16);
    __bf16* ctx  = (__bf16*)ws;  ws += pe * sizeof(__bf16);
    float*  Xln  = (float*)ws;

    const int wblocks = (int)(we / 256);
    prep_wt_kernel<<<wblocks, 256, 0, stream>>>(Wq,  WqT);
    prep_wt_kernel<<<wblocks, 256, 0, stream>>>(Wk,  WkT);
    prep_wt_kernel<<<wblocks, 256, 0, stream>>>(Wfc, WfcT);

    dim3 blk(32);
    dim3 gproj(D_MODEL / 64, M_ROWS / 32);
    proj_kernel<<<gproj, blk, 0, stream>>>(q, WqT, bq, qh, 0);
    proj_kernel<<<gproj, blk, 0, stream>>>(k, WkT, bk, kh, 0);
    proj_kernel<<<gproj, blk, 0, stream>>>(v, WkT, bk, vhT, 1);   // faithful: Wk/bk for v

    dim3 gqk(S_LEN / 64, S_LEN / 32, B_BATCH * NH);
    qk_kernel<<<gqk, blk, 0, stream>>>(qh, kh, mask, attn);

    softmax_kernel<<<B_BATCH * NH * S_LEN, 256, 0, stream>>>(attn);

    dim3 gpv(S_LEN / 16, B_BATCH * NH);
    pv_kernel<<<gpv, blk, 0, stream>>>(attn, vhT, ctx);

    fc_kernel<<<gproj, blk, 0, stream>>>(ctx, WfcT, bfc, q, Xln);

    ln_kernel<<<M_ROWS, 256, 0, stream>>>(Xln, gamma, beta, out_final);
}
